// VAMP_inverse_CircConv_net_24627342475623
// MI455X (gfx1250) — compile-verified, matrix operand order checked by reading, not run on hardware
//
#include <hip/hip_runtime.h>
#include <hip/hip_bf16.h>

typedef __attribute__((ext_vector_type(16))) __bf16 v16bf;
typedef __attribute__((ext_vector_type(8)))  __bf16 v8bf;
typedef __attribute__((ext_vector_type(8)))  float  v8f;
typedef unsigned int u32;
typedef unsigned long long u64;
typedef __attribute__((ext_vector_type(4))) u32 v4u;
typedef __attribute__((ext_vector_type(8))) int  v8i;
typedef __attribute__((ext_vector_type(4))) int  v4i;

#if __has_builtin(__builtin_amdgcn_tensor_load_to_lds)
  #define CC_HAS_TDM 1
  #if __has_include(<hip/amd_detail/amd_gfx1250_TDM.h>)
    #define CC_TDM6 1
  #endif
#endif

// Circular conv as GEMM:  M = 8*320 = 2560, N = 2*320 = 640, K = 2*320*320 = 204800
// With j' = 319 - j:
//   y[(n,i),(o,j')] = sum_{c,r,b} X2[n][c][r+i+1][b] * WT[o][c][r][b + j']
// X2 : bf16 [8][2][640][320]  (rows tiled twice; A tile = plain 2-D TDM, stride 320)
// WT : bf16 [2][2][320][640]  (cols tiled twice;  B tile = Toeplitz band, TDM stride 1)

__global__ void cc_prep_x(const float* __restrict__ x, __bf16* __restrict__ X2) {
    int idx = blockIdx.x * 256 + threadIdx.x;            // 8*2*640*320
    int b  = idx % 320;
    int a  = (idx / 320) % 640;
    int nc = idx / (320 * 640);
    int ar = (a >= 320) ? (a - 320) : a;
    X2[idx] = (__bf16)x[(nc * 320 + ar) * 320 + b];
}

__global__ void cc_prep_w(const float* __restrict__ w, __bf16* __restrict__ WT) {
    int idx = blockIdx.x * 256 + threadIdx.x;            // 2*2*320*640
    int v  = idx % 640;
    int r  = (idx / 640) % 320;
    int oc = idx / (640 * 320);
    int src = (v >= 320) ? (v - 320) : v;                // plain doubling, no reversal
    WT[idx] = (__bf16)w[(oc * 320 + r) * 320 + src];
}

#ifdef CC_HAS_TDM
// One 2-D TDM descriptor: 64 x 64 elements (2 B each), row stride in elements.
__device__ __forceinline__ void tdm_load_2d(u32 lds_addr, u64 gaddr_bytes, u32 stride_el) {
    const u32 td0 = 0x40000000u, td1 = 0x40000000u;      // huge dims: no OOB clipping
    v4u g0;
    g0[0] = 1u;                                          // count=1, user descriptor
    g0[1] = lds_addr;                                    // LDS byte address
    g0[2] = (u32)(gaddr_bytes & 0xffffffffu);            // global addr [31:0]
    g0[3] = (u32)((gaddr_bytes >> 32) & 0x01ffffffu) | (2u << 30);  // addr[56:32] | type=2
    v8i g1;
    g1[0] = (int)(1u << 16);                             // wg_mask=0, data_size=1 (2B)
    g1[1] = (int)((td0 & 0xffffu) << 16);                // tensor_dim0[15:0]
    g1[2] = (int)(((td0 >> 16) & 0xffffu) | ((td1 & 0xffffu) << 16));
    g1[3] = (int)(((td1 >> 16) & 0xffffu) | (64u << 16)); // tile_dim0 = 64
    g1[4] = (int)64;                                     // tile_dim1 = 64 (tile_dim2 = 0)
    g1[5] = (int)stride_el;                              // tensor_dim0_stride[31:0]
    g1[6] = 0;                                           // stride[47:32], dim1_stride lo
    g1[7] = 0;
    v4i gz = {0, 0, 0, 0};
#ifdef CC_TDM6
    v8i gz8 = {0, 0, 0, 0, 0, 0, 0, 0};
    __builtin_amdgcn_tensor_load_to_lds(g0, g1, gz, gz, gz8, 0);
#else
    __builtin_amdgcn_tensor_load_to_lds(g0, g1, gz, gz, 0);
#endif
}
#endif

__global__ __launch_bounds__(256) void cc_gemm_wmma(
    const __bf16* __restrict__ X2, const __bf16* __restrict__ WT,
    const float* __restrict__ x, const float* __restrict__ skipw,
    float* __restrict__ out)
{
    // WG tile: 128 (M) x 64 (N), K = 64 per double-buffered step.
    __shared__ alignas(16) __bf16 As[2][128][64];   // 32 KB
    __shared__ alignas(16) __bf16 Bs[2][64][64];    // 16 KB

    const int bm = blockIdx.x;                 // 0..19
    const int bn = blockIdx.y;                 // 0..9
    const int m_base = bm * 128;
    const int o = bn / 5, jp_base = (bn % 5) * 64;   // j' tile base

    const int t     = threadIdx.x;
    const int lane  = t & 31;
    const int wave  = t >> 5;
    const int wm    = wave >> 1;               // 0..3
    const int wn    = wave & 1;                // 0..1
    const int lm    = lane & 15;
    const int khalf = lane >> 4;

    v8f acc[2][2];
    #pragma unroll
    for (int ms = 0; ms < 2; ++ms)
        #pragma unroll
        for (int ns = 0; ns < 2; ++ns)
            acc[ms][ns] = (v8f){0.f,0.f,0.f,0.f,0.f,0.f,0.f,0.f};

    constexpr int NSTEPS = 2 * 320 * 5;        // 3200 steps of K=64
    int bcnt = 0;

#ifdef CC_HAS_TDM
    // --- per-wave TDM state, rooted in readfirstlane so it stays in SGPRs:
    //   wave 0: A rows   0..63   (stride 320)
    //   wave 1: A rows  64..127  (stride 320; may live in the next batch image n)
    //   wave 2: B Toeplitz band  (stride 1)
    const int  wv  = __builtin_amdgcn_readfirstlane(wave);        // scalar wave id
    const bool isB = (wv == 2);
    const int  mw  = m_base + ((wv == 1) ? 64 : 0);
    const int  nw  = mw / 320, iw = mw - nw * 320;
    const u64  gA  = (u64)(uintptr_t)X2 + 2ull * (((u64)(nw * 2) * 640 + (iw + 1)) * 320);
    const u64  gBb = (u64)(uintptr_t)WT + 2ull * (((u64)(o * 2) * 320) * 640 + jp_base);
    u64 gmine = isB ? gBb : gA;
    const u32 lds0 = isB ? (u32)(uintptr_t)&Bs[0][0][0]
                         : (u32)(uintptr_t)&As[0][0][0] + ((wv == 1) ? 8192u : 0u);
    const u32 lbuf  = isB ? 8192u : 16384u;    // per-buffer LDS stride
    const u32 gstr  = isB ? 1u : 320u;         // descriptor row stride (elements)
    const u32 ginc  = isB ? 768u : 128u;       // wrap-step byte increment (normal = 128)

    if (wv < 3) tdm_load_2d(lds0, gmine, gstr);
    __builtin_amdgcn_s_wait_tensorcnt(0);      // no-op for waves 3..7
    __syncthreads();
#else
    // --- fallback: cooperative per-thread loads (j-flipped, ascending, no reversal)
    const int a_row = t >> 1;
    const int a_k   = (t & 1) << 3;
    const int a_m   = m_base + a_row;
    const int a_n   = a_m / 320;
    const int a_i   = a_m - a_n * 320;
    long aoff = ((long)(a_n * 2) * 640 + (a_i + 1)) * 320 + a_k;
    const int b_j = t >> 2;
    const int b_k = (t & 3) << 3;
    long boff = ((long)(o * 2) * 320) * 640 + jp_base + b_j + b_k;
    v8bf areg[4], breg[2];
    auto gload = [&]() {
        #pragma unroll
        for (int q = 0; q < 4; ++q) areg[q] = *(const v8bf*)(X2 + aoff + q * 16);
        #pragma unroll
        for (int h = 0; h < 2; ++h) {
            #pragma unroll
            for (int e = 0; e < 8; ++e) breg[h][e] = WT[boff + h * 32 + e];
        }
    };
    auto sstore = [&](int buf) {
        #pragma unroll
        for (int q = 0; q < 4; ++q) *(v8bf*)&As[buf][a_row][a_k + q * 16] = areg[q];
        #pragma unroll
        for (int h = 0; h < 2; ++h) *(v8bf*)&Bs[buf][b_j][b_k + h * 32] = breg[h];
    };
    gload();
    sstore(0);
    __syncthreads();
#endif

    for (int s = 0; s < NSTEPS; ++s) {
        const int cur = s & 1;
        const bool more = (s + 1 < NSTEPS);

#ifdef CC_HAS_TDM
        if (more && wv < 3) {                              // scalar branch (SCC)
            const bool wrap = (bcnt == 4);
            bcnt = wrap ? 0 : bcnt + 1;
            gmine += wrap ? ginc : 128u;                   // +64 elems (B wrap: +384)
            if (s + 1 == NSTEPS / 2 && !isB) gmine += 204800u;  // A jump at c: 0 -> 1
            tdm_load_2d(lds0 + (u32)(cur ^ 1) * lbuf, gmine, gstr);
        }
#else
        if (more) {
            const bool wrap = (bcnt == 4);
            bcnt = wrap ? 0 : bcnt + 1;
            aoff += 64;
            boff += wrap ? 384 : 64;
            if (s + 1 == NSTEPS / 2) aoff += 102400;
            gload();
        }
#endif

        // Per K-chunk fragment scope (low register pressure -> in-place C/D coalescing).
        #pragma unroll
        for (int kx = 0; kx < 2; ++kx) {
            const int kk = kx * 32 + (khalf << 3);
            v16bf af[2], bf[2];
            #pragma unroll
            for (int ms = 0; ms < 2; ++ms) {
                const __bf16* p = &As[cur][(wm << 5) + (ms << 4) + lm][kk];
                v8bf lo = *(const v8bf*)p;
                v8bf hi = *(const v8bf*)(p + 16);
                af[ms] = __builtin_shufflevector(lo, hi,
                    0,1,2,3,4,5,6,7,8,9,10,11,12,13,14,15);
            }
            #pragma unroll
            for (int ns = 0; ns < 2; ++ns) {
                const __bf16* p = &Bs[cur][(wn << 5) + (ns << 4) + lm][kk];
                v8bf lo = *(const v8bf*)p;
                v8bf hi = *(const v8bf*)(p + 16);
                bf[ns] = __builtin_shufflevector(lo, hi,
                    0,1,2,3,4,5,6,7,8,9,10,11,12,13,14,15);
            }
            #pragma unroll
            for (int ms = 0; ms < 2; ++ms)
                #pragma unroll
                for (int ns = 0; ns < 2; ++ns)
                    acc[ms][ns] = __builtin_amdgcn_wmma_f32_16x16x32_bf16(
                        false, af[ms], false, bf[ns], (short)0,
                        acc[ms][ns], false, false);
        }

#ifdef CC_HAS_TDM
        __builtin_amdgcn_s_wait_tensorcnt(0);         // no-op for waves 3..7
#else
        if (more) sstore(cur ^ 1);
#endif
        __syncthreads();
    }

    // epilogue: out = skip_w * x + acc ; output column j = 319 - j'
    const float sw = skipw[0];
    #pragma unroll
    for (int ms = 0; ms < 2; ++ms) {
        #pragma unroll
        for (int ns = 0; ns < 2; ++ns) {
            const int jp = jp_base + (wn << 5) + (ns << 4) + lm;
            const int j  = 319 - jp;
            #pragma unroll
            for (int v = 0; v < 8; ++v) {
                const int m = m_base + (wm << 5) + (ms << 4) + v + (khalf << 3);
                const int n = m / 320;
                const int i = m - n * 320;
                const long oidx = ((long)(n * 640 + o * 320 + i)) * 320 + j;
                out[oidx] = sw * x[oidx] + acc[ms][ns][v];
            }
        }
    }
}

extern "C" void kernel_launch(void* const* d_in, const int* in_sizes, int n_in,
                              void* d_out, int out_size, void* d_ws, size_t ws_size,
                              hipStream_t stream) {
    const float* x  = (const float*)d_in[0];
    const float* w  = (const float*)d_in[1];
    const float* sw = (const float*)d_in[2];
    float* out = (float*)d_out;

    __bf16* X2 = (__bf16*)d_ws;                                          // 6,553,600 B
    __bf16* WT = (__bf16*)((char*)d_ws + (size_t)8 * 2 * 640 * 320 * 2); // +1,638,400 B

    cc_prep_x<<<(8 * 2 * 640 * 320) / 256, 256, 0, stream>>>(x, X2);
    cc_prep_w<<<(2 * 2 * 320 * 640) / 256, 256, 0, stream>>>(w, WT);

    dim3 grid(20, 10);   // M tiles x N tiles
    cc_gemm_wmma<<<grid, 256, 0, stream>>>(X2, WT, x, sw, out);
}
